// debugModel_3487513444611
// MI455X (gfx1250) — matrix operand order repaired and checked by reading, not compiled, not run on hardware
//
#include <hip/hip_runtime.h>

// CDNA5 (gfx1250) fused GNN link-prediction head.
// Reference's graph aggregation is an identity (message = h[dst] mean-reduced
// at dst => h_new == h), so:
//   out = relu(nf[head]W1+b1) @ Wp[:512] + relu(nf[tail]W1+b1) @ Wp[512:] + bp
// Computed with V_WMMA_F32_16X16X32_BF16 using the bf16x3 split
// (aH*bH + aH*bL + aL*bH, fp32 accumulate) for fp32-class accuracy at bf16
// matrix-core rate (8x the FLOP/instruction of the f32 K=4 WMMA).

typedef __bf16 v2bf  __attribute__((ext_vector_type(2)));
typedef __bf16 v8bf  __attribute__((ext_vector_type(8)));
typedef __bf16 v16bf __attribute__((ext_vector_type(16)));
typedef float  v8f   __attribute__((ext_vector_type(8)));

#define NODE_IN   768
#define NODE_D    512
#define OUT_N     97
#define MROWS     16384      // B * P
#define ROWS      128        // output rows per block
#define BK        32         // K chunk (== bf16 WMMA K)
#define CC        64         // stage-1 column chunk (stage-2 K chunk, 2 halves)
#define NT_OUT    7          // 7*16 = 112 >= 97 padded output columns
#define KS        40         // LDS row stride in bf16 (80B = 5*16B: aligned b128
                             // fragment loads, conflict-free across 16 rows)

// LDS offsets in bf16 elements (total 25600 bf16 = 51200 B + 512 B indices).
// lWp aliases lA (disjoint lifetimes, barrier-separated).
#define OFF_AH    0          // [128][KS] stage-1 A hi
#define OFF_AL    5120       // [128][KS] stage-1 A lo
#define OFF_WTH   10240      // [64][KS]  W1 chunk, transposed, hi
#define OFF_WTL   12800      // [64][KS]  W1 chunk, transposed, lo
#define OFF_HH    15360      // [128][KS] relu(h) chunk half, hi
#define OFF_HL    20480      // [128][KS] relu(h) chunk half, lo
#define OFF_WPH   0          // [112][KS] Wp chunk, transposed, hi (aliases lA)
#define OFF_WPL   4480       // [112][KS] Wp chunk, transposed, lo
#define SMEM_BF16 25600

__device__ __forceinline__ void cvt_split(float x, __bf16& h, __bf16& l) {
    h = (__bf16)x;                    // RNE
    l = (__bf16)(x - (float)h);       // residual
}

// A fragment 16x32 bf16: lane (l16,kh) holds K in [8kh,8kh+8) and [16+8kh,16+8kh+8)
__device__ __forceinline__ v16bf load_a_frag(const __bf16* rowp, int kh) {
    v8bf p0 = *(const v8bf*)(rowp + 8 * kh);
    v8bf p1 = *(const v8bf*)(rowp + 16 + 8 * kh);
    return __builtin_shufflevector(p0, p1, 0,1,2,3,4,5,6,7,8,9,10,11,12,13,14,15);
}
// B fragment 32x16 bf16: lane (l16,kh) holds column l16, K in [8kh,8kh+8) and [16+8kh,16+8kh+8)
__device__ __forceinline__ v16bf load_b_frag(const __bf16* colp, int kh) {
    v8bf q0 = *(const v8bf*)(colp + 8 * kh);
    v8bf q1 = *(const v8bf*)(colp + 16 + 8 * kh);
    return __builtin_shufflevector(q0, q1, 0,1,2,3,4,5,6,7,8,9,10,11,12,13,14,15);
}

__device__ __forceinline__ v8f wmma3(v16bf aH, v16bf aL, v16bf bH, v16bf bL, v8f c) {
    c = __builtin_amdgcn_wmma_f32_16x16x32_bf16(false, aH, false, bH, (short)0, c, false, false);
    c = __builtin_amdgcn_wmma_f32_16x16x32_bf16(false, aH, false, bL, (short)0, c, false, false);
    c = __builtin_amdgcn_wmma_f32_16x16x32_bf16(false, aL, false, bH, (short)0, c, false, false);
    return c;
}

__global__ __launch_bounds__(256, 1)
void fused_gnn(const float* __restrict__ nf,      // [50000][768]
               const float* __restrict__ W1,      // [768][512]
               const float* __restrict__ b1,      // [512]
               const float* __restrict__ Wp,      // [1024][97]
               const float* __restrict__ bp,      // [97]
               const int*   __restrict__ headIdx, // [16384]
               const int*   __restrict__ tailIdx, // [16384]
               float*       __restrict__ out)     // [16384][97]
{
    __shared__ alignas(16) __bf16 smem[SMEM_BF16];
    __shared__ int lIdx[ROWS];

    __bf16* lAh  = smem + OFF_AH;
    __bf16* lAl  = smem + OFF_AL;
    __bf16* lWth = smem + OFF_WTH;
    __bf16* lWtl = smem + OFF_WTL;
    __bf16* lHh  = smem + OFF_HH;
    __bf16* lHl  = smem + OFF_HL;
    __bf16* lWph = smem + OFF_WPH;
    __bf16* lWpl = smem + OFF_WPL;

    const int t    = threadIdx.x;
    const int lane = t & 31;
    const int wv   = t >> 5;        // wave 0..7, rows wv*16 .. wv*16+15
    const int kh   = lane >> 4;     // half-wave select
    const int l16  = lane & 15;
    const int rowBase = blockIdx.x * ROWS;

    v8f outAcc[NT_OUT];
    #pragma unroll
    for (int i = 0; i < NT_OUT; ++i)
        outAcc[i] = (v8f){0.f,0.f,0.f,0.f,0.f,0.f,0.f,0.f};

    #pragma unroll 1
    for (int part = 0; part < 2; ++part) {
        const int* __restrict__ idxp = (part == 0) ? headIdx : tailIdx;
        __syncthreads();
        if (t < ROWS) lIdx[t] = idxp[rowBase + t];
        __syncthreads();

        #pragma unroll 1
        for (int cc = 0; cc < NODE_D; cc += CC) {
            // ============ Stage 1: h = relu(nf[idx] @ W1[:, cc:cc+64] + b1) ======
            v8f acc[4];
            #pragma unroll
            for (int i = 0; i < 4; ++i)
                acc[i] = (v8f){0.f,0.f,0.f,0.f,0.f,0.f,0.f,0.f};

            #pragma unroll 1
            for (int k0 = 0; k0 < NODE_IN; k0 += BK) {
                __syncthreads();   // lA/lWt (and aliased lWp) free to overwrite
                // gather 128 nf rows x 32 K, split fp32 -> bf16 hi/lo
                #pragma unroll
                for (int j = 0; j < 8; ++j) {
                    const int idx = j * 256 + t;      // 0..2047 element-pairs
                    const int r = idx >> 4, p = idx & 15;
                    const float2 x = *(const float2*)(nf + (size_t)lIdx[r] * NODE_IN + k0 + 2 * p);
                    __bf16 h0, l0, h1, l1;
                    cvt_split(x.x, h0, l0); cvt_split(x.y, h1, l1);
                    v2bf hv = {h0, h1}, lv = {l0, l1};
                    *(v2bf*)&lAh[r * KS + 2 * p] = hv;
                    *(v2bf*)&lAl[r * KS + 2 * p] = lv;
                }
                // W1 chunk [32][64] -> transposed [n][k], bf16 hi/lo
                #pragma unroll
                for (int j = 0; j < 4; ++j) {
                    const int idx = j * 256 + t;      // 0..1023 pairs
                    const int n = idx & 63, kp = idx >> 6;   // kp 0..15
                    const float x0 = W1[(size_t)(k0 + 2 * kp)     * NODE_D + cc + n];
                    const float x1 = W1[(size_t)(k0 + 2 * kp + 1) * NODE_D + cc + n];
                    __bf16 h0, l0, h1, l1;
                    cvt_split(x0, h0, l0); cvt_split(x1, h1, l1);
                    v2bf hv = {h0, h1}, lv = {l0, l1};
                    *(v2bf*)&lWth[n * KS + 2 * kp] = hv;
                    *(v2bf*)&lWtl[n * KS + 2 * kp] = lv;
                }
                __syncthreads();

                const __bf16* arow = lAh + (wv * 16 + l16) * KS;
                const __bf16* arol = lAl + (wv * 16 + l16) * KS;
                const v16bf aH = load_a_frag(arow, kh);
                const v16bf aL = load_a_frag(arol, kh);
                #pragma unroll
                for (int tt = 0; tt < 4; ++tt) {       // 16 rows x 64 cols per wave
                    const int n = tt * 16 + l16;
                    const v16bf bH = load_b_frag(lWth + n * KS, kh);
                    const v16bf bL = load_b_frag(lWtl + n * KS, kh);
                    acc[tt] = wmma3(aH, aL, bH, bL, acc[tt]);
                }
            }

            // ============ Stage 2 in two K=32 halves (reuses small LDS) ==========
            #pragma unroll 1
            for (int half = 0; half < 2; ++half) {
                __syncthreads();   // prior lH/lWp readers (and stage-1 lA readers) done
                // write relu(h)+b1 half -> lH as bf16 hi/lo
                #pragma unroll
                for (int tl = 0; tl < 2; ++tl) {
                    const int tt = 2 * half + tl;
                    const float bias = b1[cc + tt * 16 + l16];
                    #pragma unroll
                    for (int v = 0; v < 8; ++v) {
                        const float x = fmaxf(acc[tt][v] + bias, 0.0f);
                        __bf16 h, l;
                        cvt_split(x, h, l);
                        const int off = (wv * 16 + v + 8 * kh) * KS + tl * 16 + l16;
                        lHh[off] = h; lHl[off] = l;
                    }
                }
                // Wp rows [kb2, kb2+32) x cols [0,112) -> transposed bf16 hi/lo
                const int kb2 = part * NODE_D + cc + 32 * half;
                #pragma unroll
                for (int j = 0; j < 7; ++j) {
                    const int idx = j * 256 + t;       // 0..1791 pairs
                    const int n = idx % 112, kp = idx / 112;   // kp 0..15
                    float x0 = 0.0f, x1 = 0.0f;
                    if (n < OUT_N) {
                        x0 = Wp[(size_t)(kb2 + 2 * kp)     * OUT_N + n];
                        x1 = Wp[(size_t)(kb2 + 2 * kp + 1) * OUT_N + n];
                    }
                    __bf16 h0, l0, h1, l1;
                    cvt_split(x0, h0, l0); cvt_split(x1, h1, l1);
                    v2bf hv = {h0, h1}, lv = {l0, l1};
                    *(v2bf*)&lWph[n * KS + 2 * kp] = hv;
                    *(v2bf*)&lWpl[n * KS + 2 * kp] = lv;
                }
                __syncthreads();

                const v16bf aH = load_a_frag(lHh + (wv * 16 + l16) * KS, kh);
                const v16bf aL = load_a_frag(lHl + (wv * 16 + l16) * KS, kh);
                #pragma unroll
                for (int tt = 0; tt < NT_OUT; ++tt) {
                    const int n = tt * 16 + l16;
                    const v16bf bH = load_b_frag(lWph + n * KS, kh);
                    const v16bf bL = load_b_frag(lWpl + n * KS, kh);
                    outAcc[tt] = wmma3(aH, aL, bH, bL, outAcc[tt]);
                }
            }
        }
    }

    // ============ Epilogue: + bp, masked store (112 -> 97 cols) =================
    #pragma unroll
    for (int tt = 0; tt < NT_OUT; ++tt) {
        const int col = tt * 16 + l16;
        if (col < OUT_N) {
            const float bias = bp[col];
            #pragma unroll
            for (int v = 0; v < 8; ++v) {
                const int r = rowBase + wv * 16 + v + 8 * kh;
                out[(size_t)r * OUT_N + col] = outAcc[tt][v] + bias;
            }
        }
    }
}

extern "C" void kernel_launch(void* const* d_in, const int* in_sizes, int n_in,
                              void* d_out, int out_size, void* d_ws, size_t ws_size,
                              hipStream_t stream) {
    (void)in_sizes; (void)n_in; (void)out_size; (void)d_ws; (void)ws_size;
    const float* nf   = (const float*)d_in[0];  // node_features
    // d_in[1] edge_features, d_in[6] src, d_in[7] dst unused: aggregation is identity
    const float* W1   = (const float*)d_in[2];
    const float* b1   = (const float*)d_in[3];
    const float* Wp   = (const float*)d_in[4];
    const float* bp   = (const float*)d_in[5];
    const int*   head = (const int*)d_in[8];
    const int*   tail = (const int*)d_in[9];
    float* out = (float*)d_out;

    dim3 grid(MROWS / ROWS);   // 128 blocks
    dim3 block(256);           // 8 waves (wave32)
    fused_gnn<<<grid, block, 0, stream>>>(nf, W1, b1, Wp, bp, head, tail, out);
}
